// MoEFFNLayer_23407571763646
// MI455X (gfx1250) — matrix operand order repaired and checked by reading, not compiled, not run on hardware
//
#include <hip/hip_runtime.h>

#define N_TOK 8192
#define DD 1024
#define HH 4096
#define NE 8
#define TILE_M 32
#define HC 128      // H-chunk per fused stage
#define KS 128      // k-superstep staged through LDS for GEMM1's B
#define XSTR 1032   // x_tile row stride (2064B: 16B aligned, conflict-free)
#define HSTR 136    // h_tile row stride (272B: 16B aligned)
#define TSTR1 136   // w1^T tile row stride (272B: 16B aligned, 68-dword stride)
#define TSTR2 40    // w2^T tile row stride (80B: 16B aligned, 20-dword stride)

typedef __attribute__((ext_vector_type(16))) __bf16 v16bf;
typedef __attribute__((ext_vector_type(8)))  float  v8f;

static __device__ __forceinline__ unsigned pack_bf16(float lo, float hi) {
  unsigned a = (unsigned)__builtin_bit_cast(unsigned short, (__bf16)lo);
  unsigned b = (unsigned)__builtin_bit_cast(unsigned short, (__bf16)hi);
  return (b << 16) | a;
}

// A fragment (16x32 bf16, M x K) from LDS [m][k]: one contiguous 16-elem run
// per lane -> 2x ds_load_b128 when the row stride is 16B aligned.
static __device__ __forceinline__ v16bf load_a_frag(const __bf16* base, int stride_e, int lane) {
  const __bf16* p = base + (lane & 15) * stride_e + ((lane >> 4) << 4);
  v16bf a;
#pragma unroll
  for (int i = 0; i < 16; ++i) a[i] = p[i];
  return a;
}

// B fragment (32x16 bf16, K x N) from a TRANSPOSED LDS tile [n][k]:
// one contiguous 16-elem run per lane -> 2x ds_load_b128.
static __device__ __forceinline__ v16bf load_bT_frag(const __bf16* rowbase, int stride_e, int lane) {
  const __bf16* p = rowbase + (lane & 15) * stride_e + ((lane >> 4) << 4);
  v16bf b;
#pragma unroll
  for (int i = 0; i < 16; ++i) b[i] = p[i];
  return b;
}

__global__ void zero_kernel(float* __restrict__ out, int n, int* __restrict__ counts) {
  int i = blockIdx.x * blockDim.x + threadIdx.x;
  if (i < n) out[i] = 0.0f;
  if (i < NE) counts[i] = 0;
}

__global__ __launch_bounds__(256) void router_kernel(
    const float* __restrict__ xs, const float* __restrict__ gate_w,
    int* __restrict__ counts, int* __restrict__ tok_idx, float* __restrict__ tok_wt) {
  __shared__ float g[NE * DD];   // 32 KB
  int tid = threadIdx.x;
  for (int i = tid; i < NE * DD; i += 256) g[i] = gate_w[i];
  __syncthreads();

  int n = blockIdx.x * 256 + tid;
  float acc[NE];
#pragma unroll
  for (int e = 0; e < NE; ++e) acc[e] = 0.0f;
  const float* xrow = xs + (long)n * DD;
  for (int d = 0; d < DD; d += 4) {
    float4 xv = *(const float4*)(xrow + d);
#pragma unroll
    for (int e = 0; e < NE; ++e) {
      const float* gr = g + e * DD + d;
      acc[e] = fmaf(xv.x, gr[0], acc[e]);
      acc[e] = fmaf(xv.y, gr[1], acc[e]);
      acc[e] = fmaf(xv.z, gr[2], acc[e]);
      acc[e] = fmaf(xv.w, gr[3], acc[e]);
    }
  }
  int e0 = 0; float l0 = acc[0];
#pragma unroll
  for (int e = 1; e < NE; ++e) if (acc[e] > l0) { l0 = acc[e]; e0 = e; }
  int e1 = (e0 == 0) ? 1 : 0; float l1 = acc[e1];
#pragma unroll
  for (int e = 0; e < NE; ++e) if (e != e0 && acc[e] > l1) { l1 = acc[e]; e1 = e; }
  float p1  = __expf(l1 - l0);
  float inv = 1.0f / (1.0f + p1);
  float w0 = inv, w1v = p1 * inv;

  int p = atomicAdd(&counts[e0], 1);
  tok_idx[e0 * N_TOK + p] = n;  tok_wt[e0 * N_TOK + p] = w0;
  p = atomicAdd(&counts[e1], 1);
  tok_idx[e1 * N_TOK + p] = n;  tok_wt[e1 * N_TOK + p] = w1v;
}

__global__ __launch_bounds__(256) void moe_ffn_kernel(
    const float* __restrict__ xs, const float* __restrict__ w1,
    const float* __restrict__ w2, const int* __restrict__ counts,
    const int* __restrict__ tok_idx, const float* __restrict__ tok_wt,
    float* __restrict__ out) {
  const int e   = blockIdx.y;
  const int cnt = counts[e];
  const int m0  = blockIdx.x * TILE_M;
  if (m0 >= cnt) return;

  const int tid  = threadIdx.x;
  const int lane = tid & 31;
  const int wave = tid >> 5;            // 8 waves

  __shared__ __bf16 x_tile[TILE_M * XSTR];   // 66.0 KB, [m][k]
  __shared__ __bf16 w1_t[HC * TSTR1];        // 34.8 KB, [h-col][k] (transposed)
  __shared__ __bf16 w2_t[512 * TSTR2];       // 40.9 KB, [d-col][k] (transposed)
  __shared__ __bf16 h_tile[TILE_M * HSTR];   //  8.7 KB, [m][k]
  __shared__ int    s_tok[TILE_M];
  __shared__ float  s_wt[TILE_M];

  if (tid < TILE_M) {
    int r  = m0 + tid;
    int rc = (r < cnt) ? r : (cnt - 1);
    s_tok[tid] = tok_idx[(long)e * N_TOK + rc];
    s_wt[tid]  = (r < cnt) ? tok_wt[(long)e * N_TOK + rc] : 0.0f;
  }
  __syncthreads();

  // Gather + convert the 32 token rows of x into LDS (b128 loads, packed cvt).
  for (int i = tid * 4; i < TILE_M * DD; i += 256 * 4) {
    int row = i >> 10;
    int col = i & (DD - 1);
    const float4 xv = *(const float4*)(xs + (long)s_tok[row] * DD + col);
    __bf16* dst = x_tile + row * XSTR + col;
    *(unsigned*)(dst)     = pack_bf16(xv.x, xv.y);
    *(unsigned*)(dst + 2) = pack_bf16(xv.z, xv.w);
  }
  __syncthreads();

  const float* w1e = w1 + (long)e * DD * HH;   // [d][h], ld = HH
  const float* w2e = w2 + (long)e * HH * DD;   // [h][d], ld = DD

  v8f acc[2][8] = {};   // [m-tile][half*4+t] : wave's 32 x 128 out slice

#pragma unroll 1
  for (int hc = 0; hc < HH; hc += HC) {
    // ============ GEMM1: h[32 x 16] slice (cols hc + wave*16), K = D =========
    v8f hacc0 = {}, hacc1 = {};
#pragma unroll 1
    for (int ks = 0; ks < DD; ks += KS) {
      __syncthreads();
      // stage w1[ks..ks+128, hc..hc+128] transposed -> w1_t[col][k]
#pragma unroll
      for (int p = 0; p < 8; ++p) {
        int pi   = tid + p * 256;            // 0..2047
        int col4 = (pi & 31) << 2;
        int k    = (pi >> 5) << 1;
        const float4 v0 = *(const float4*)(w1e + (long)(ks + k)     * HH + hc + col4);
        const float4 v1 = *(const float4*)(w1e + (long)(ks + k + 1) * HH + hc + col4);
        __bf16* dst = w1_t + col4 * TSTR1 + k;
        *(unsigned*)(dst)             = pack_bf16(v0.x, v1.x);
        *(unsigned*)(dst + TSTR1)     = pack_bf16(v0.y, v1.y);
        *(unsigned*)(dst + 2 * TSTR1) = pack_bf16(v0.z, v1.z);
        *(unsigned*)(dst + 3 * TSTR1) = pack_bf16(v0.w, v1.w);
      }
      __syncthreads();
#pragma unroll
      for (int kk = 0; kk < KS; kk += 32) {
        v16bf b  = load_bT_frag(w1_t + (wave * 16) * TSTR1 + kk, TSTR1, lane);
        v16bf a0 = load_a_frag(x_tile + ks + kk, XSTR, lane);
        v16bf a1 = load_a_frag(x_tile + 16 * XSTR + ks + kk, XSTR, lane);
        hacc0 = __builtin_amdgcn_wmma_f32_16x16x32_bf16(false, a0, false, b,
                                                        (short)0, hacc0, false, false);
        hacc1 = __builtin_amdgcn_wmma_f32_16x16x32_bf16(false, a1, false, b,
                                                        (short)0, hacc1, false, false);
      }
    }
    __syncthreads();   // previous chunk's h_tile readers done
    {
      int n  = lane & 15;
      int mb = (lane >> 4) << 3;
#pragma unroll
      for (int j = 0; j < 8; ++j) {
        float r0 = hacc0[j];
        float r1 = hacc1[j];
        r0 = (r0 > 0.0f) ? r0 : 0.0f;            // relu
        r1 = (r1 > 0.0f) ? r1 : 0.0f;
        h_tile[(mb + j) * HSTR + wave * 16 + n]        = (__bf16)r0;
        h_tile[(16 + mb + j) * HSTR + wave * 16 + n]   = (__bf16)r1;
      }
    }
    // ===== GEMM2: out[32 x 1024] += h[32 x 128] * w2[hc..+128, :], 2 halves ==
#pragma unroll
    for (int half = 0; half < 2; ++half) {
      const int c0 = half * 512;
#pragma unroll 1
      for (int kk = 0; kk < HC; kk += 32) {
        __syncthreads();
        // stage w2[hc+kk..+32, c0..c0+512] transposed -> w2_t[col][k]
#pragma unroll
        for (int p = 0; p < 8; ++p) {
          int pi   = tid + p * 256;          // 0..2047
          int col4 = (pi & 127) << 2;
          int k    = (pi >> 7) << 1;
          const float4 v0 = *(const float4*)(w2e + (long)(hc + kk + k)     * DD + c0 + col4);
          const float4 v1 = *(const float4*)(w2e + (long)(hc + kk + k + 1) * DD + c0 + col4);
          __bf16* dst = w2_t + col4 * TSTR2 + k;
          *(unsigned*)(dst)             = pack_bf16(v0.x, v1.x);
          *(unsigned*)(dst + TSTR2)     = pack_bf16(v0.y, v1.y);
          *(unsigned*)(dst + 2 * TSTR2) = pack_bf16(v0.z, v1.z);
          *(unsigned*)(dst + 3 * TSTR2) = pack_bf16(v0.w, v1.w);
        }
        __syncthreads();
        v16bf a0 = load_a_frag(h_tile + kk, HSTR, lane);
        v16bf a1 = load_a_frag(h_tile + 16 * HSTR + kk, HSTR, lane);
#pragma unroll
        for (int t = 0; t < 4; ++t) {
          v16bf b = load_bT_frag(w2_t + (wave * 64 + t * 16) * TSTR2, TSTR2, lane);
          acc[0][half * 4 + t] = __builtin_amdgcn_wmma_f32_16x16x32_bf16(
              false, a0, false, b, (short)0, acc[0][half * 4 + t], false, false);
          acc[1][half * 4 + t] = __builtin_amdgcn_wmma_f32_16x16x32_bf16(
              false, a1, false, b, (short)0, acc[1][half * 4 + t], false, false);
        }
      }
    }
  }

  // ---- epilogue: scale by combine weight, atomically accumulate into out
  {
    int n  = lane & 15;
    int mb = (lane >> 4) << 3;
#pragma unroll
    for (int r = 0; r < 2; ++r) {
#pragma unroll
      for (int j = 0; j < 8; ++j) {
        int m = r * 16 + mb + j;
        if (m0 + m < cnt) {
          long  obase = (long)s_tok[m] * DD;
          float wgt   = s_wt[m];
#pragma unroll
          for (int a8 = 0; a8 < 8; ++a8) {
            int col = ((a8 >> 2) * 512) + wave * 64 + ((a8 & 3) * 16) + n;
            atomicAdd(out + obase + col, acc[r][a8][j] * wgt);
          }
        }
      }
    }
  }
}

extern "C" void kernel_launch(void* const* d_in, const int* in_sizes, int n_in,
                              void* d_out, int out_size, void* d_ws, size_t ws_size,
                              hipStream_t stream) {
  const float* xs     = (const float*)d_in[0];
  const float* gate_w = (const float*)d_in[1];
  const float* w1     = (const float*)d_in[2];
  const float* w2     = (const float*)d_in[3];
  float* out = (float*)d_out;

  // scratch layout: counts[32] | tok_idx[E*N] | tok_wt[E*N]   (~512 KB)
  int*   counts  = (int*)d_ws;
  int*   tok_idx = counts + 32;
  float* tok_wt  = (float*)(tok_idx + NE * N_TOK);

  zero_kernel<<<(N_TOK * DD + 255) / 256, 256, 0, stream>>>(out, N_TOK * DD, counts);
  router_kernel<<<N_TOK / 256, 256, 0, stream>>>(xs, gate_w, counts, tok_idx, tok_wt);
  dim3 grid(N_TOK / TILE_M, NE);
  moe_ffn_kernel<<<grid, 256, 0, stream>>>(xs, w1, w2, counts, tok_idx, tok_wt, out);
}
